// Attention_79173427134629
// MI455X (gfx1250) — matrix operand order repaired and checked by reading, not compiled, not run on hardware
//
#include <hip/hip_runtime.h>

typedef __attribute__((ext_vector_type(16))) __bf16 v16bf;
typedef __attribute__((ext_vector_type(8)))  __bf16 v8bf;
typedef __attribute__((ext_vector_type(8)))  float  v8f;

namespace {
constexpr int kB  = 32;
constexpr int kN  = 577;
constexpr int kC  = 768;
constexpr int kH  = 12;
constexpr int kD  = 64;
constexpr int kC3 = 3 * kC;             // 2304
constexpr int kNPad = 608;              // 19 * 32 (>= 577, multiple of 32)
constexpr int kRows = kB * kN;          // 18464 (= 577 * 32, exact)
constexpr int kRowTiles32 = kRows / 32; // 577
constexpr int kJT = kNPad / 16;         // 38 column tiles for S
constexpr int kKTpv = kNPad / 32;       // 19 K-steps for P*V
}

// ---------------------------------------------------------------- helpers

__device__ __forceinline__ __bf16 f2bf(float f) {
  // round-to-nearest-even f32 -> bf16
  unsigned u = __float_as_uint(f);
  u += 0x7FFFu + ((u >> 16) & 1u);
  unsigned short h = (unsigned short)(u >> 16);
  __bf16 r;
  __builtin_memcpy(&r, &h, sizeof(r));
  return r;
}

__device__ __forceinline__ v8f wmma_bf16(v16bf a, v16bf b, v8f c) {
  return __builtin_amdgcn_wmma_f32_16x16x32_bf16(
      /*neg_a=*/false, a, /*neg_b=*/false, b,
      /*c_mod=*/(short)0, c, /*reuse_a=*/false, /*reuse_b=*/false);
}

// A-fragment (16x32 bf16, lane = row M, two contiguous 8-elem K chunks).
// p points at row_base + k0 + 8*g ; second chunk at +16 elements.
__device__ __forceinline__ v16bf load_a_frag(const __bf16* p) {
  v8bf c0 = *reinterpret_cast<const v8bf*>(p);
  v8bf c1 = *reinterpret_cast<const v8bf*>(p + 16);
  v16bf a;
#pragma unroll
  for (int e = 0; e < 8; ++e) { a[e] = c0[e]; a[e + 8] = c1[e]; }
  return a;
}

__device__ __forceinline__ v16bf load_b_frag(const __bf16* p) {
  return *reinterpret_cast<const v16bf*>(p);
}

// ---------------------------------------------------------------- stage 0

__global__ void cvt_f32_bf16_kernel(const float* __restrict__ src,
                                    __bf16* __restrict__ dst, int n) {
  int stride = gridDim.x * blockDim.x;
  for (int i = blockIdx.x * blockDim.x + threadIdx.x; i < n; i += stride)
    dst[i] = f2bf(src[i]);
}

// Pre-swizzle a row-major f32 weight (K x ncols) into B-fragment order:
// frag (kt,nt) holds 32x16, lane holds col nt*16+(lane&15),
// elements e = K values kt*32 + 8*(lane>>4) + (e&7) + 16*(e>>3) (same order as load_a_frag).
__global__ void swizzle_w_kernel(const float* __restrict__ w,
                                 __bf16* __restrict__ wf, int ncols, int total) {
  int stride = gridDim.x * blockDim.x;
  int ntiles = ncols >> 4;
  for (int f = blockIdx.x * blockDim.x + threadIdx.x; f < total; f += stride) {
    int e    = f & 15;
    int lane = (f >> 4) & 31;
    int frag = f >> 9;
    int nt = frag % ntiles;
    int kt = frag / ntiles;
    int krow = kt * 32 + ((lane >> 4) << 3) + (e & 7) + ((e >> 3) << 4);
    int ncol = nt * 16 + (lane & 15);
    wf[f] = f2bf(w[krow * ncols + ncol]);
  }
}

// ---------------------------------------------------------------- stage 1: QKV GEMM
// Per wave: 32(M) x 64(N) tile, K = 768, software-pipelined (double-buffered
// register fragments so next-step loads overlap current-step WMMAs).

__global__ void __launch_bounds__(256)
qkv_gemm_kernel(const __bf16* __restrict__ xb, const __bf16* __restrict__ wf,
                const float* __restrict__ bias,
                __bf16* __restrict__ q, __bf16* __restrict__ kk,
                __bf16* __restrict__ vT) {
  const int lane = threadIdx.x & 31;
  const int wave = threadIdx.x >> 5;
  constexpr int colTiles64 = kC3 / 64;  // 36
  constexpr int ntilesB = kC3 / 16;     // 144
  constexpr int KT = kC / 32;           // 24
  const int tile = blockIdx.x * 8 + wave;
  if (tile >= kRowTiles32 * colTiles64) return;
  const int rt = tile / colTiles64;
  const int ct = tile % colTiles64;
  const int m = lane & 15, g = lane >> 4;

  const __bf16* arow0 = xb + (rt * 32 + m) * kC + 8 * g;
  const __bf16* arow1 = xb + (rt * 32 + 16 + m) * kC + 8 * g;
  const __bf16* bbase = wf + ((ct * 4) << 9) + (lane << 4);

  v8f acc[2][4];
#pragma unroll
  for (int i = 0; i < 2; ++i)
#pragma unroll
    for (int j = 0; j < 4; ++j) acc[i][j] = v8f{};

  // prologue: fragments for kt = 0
  v16bf a0 = load_a_frag(arow0);
  v16bf a1 = load_a_frag(arow1);
  v16bf b0 = load_b_frag(bbase);
  v16bf b1 = load_b_frag(bbase + 512);
  v16bf b2 = load_b_frag(bbase + 1024);
  v16bf b3 = load_b_frag(bbase + 1536);

  for (int kt = 0; kt < KT; ++kt) {
    const int ktn = (kt + 1 < KT) ? kt + 1 : kt;   // clamped (last iter reloads)
    // issue next-step loads first so they overlap this step's WMMAs
    v16bf na0 = load_a_frag(arow0 + ktn * 32);
    v16bf na1 = load_a_frag(arow1 + ktn * 32);
    const __bf16* nb = bbase + ((ktn * ntilesB) << 9);
    __builtin_prefetch(nb + (ntilesB << 9), 0, 1);
    v16bf nb0 = load_b_frag(nb);
    v16bf nb1 = load_b_frag(nb + 512);
    v16bf nb2 = load_b_frag(nb + 1024);
    v16bf nb3 = load_b_frag(nb + 1536);

    acc[0][0] = wmma_bf16(a0, b0, acc[0][0]);
    acc[0][1] = wmma_bf16(a0, b1, acc[0][1]);
    acc[0][2] = wmma_bf16(a0, b2, acc[0][2]);
    acc[0][3] = wmma_bf16(a0, b3, acc[0][3]);
    acc[1][0] = wmma_bf16(a1, b0, acc[1][0]);
    acc[1][1] = wmma_bf16(a1, b1, acc[1][1]);
    acc[1][2] = wmma_bf16(a1, b2, acc[1][2]);
    acc[1][3] = wmma_bf16(a1, b3, acc[1][3]);

    a0 = na0; a1 = na1; b0 = nb0; b1 = nb1; b2 = nb2; b3 = nb3;
  }

#pragma unroll
  for (int j = 0; j < 4; ++j) {
    const int c = ct * 64 + j * 16 + m;   // column in [0, 2304)
    const float bv = bias[c];
    const int which = c / kC;
    const int cc = c % kC;
    const int h = cc >> 6, d = cc & 63;
#pragma unroll
    for (int i = 0; i < 2; ++i) {
#pragma unroll
      for (int r = 0; r < 8; ++r) {
        int row = rt * 32 + i * 16 + 8 * g + r;  // flat b*N + n
        int b = row / kN, n = row % kN;
        int bh = b * kH + h;
        float val = acc[i][j][r] + bv;
        if (which == 0)      q [(bh * kNPad + n) * kD + d] = f2bf(val);
        else if (which == 1) kk[(bh * kNPad + n) * kD + d] = f2bf(val);
        else                 vT[(bh * kD + d) * kNPad + n] = f2bf(val);  // transposed
      }
    }
  }
}

// ---------------------------------------------------------------- stage 2: fused attention

__global__ void __launch_bounds__(128)
attn_kernel(const __bf16* __restrict__ q, const __bf16* __restrict__ kk,
            const __bf16* __restrict__ vT, const float* __restrict__ scale,
            __bf16* __restrict__ ao) {
  __shared__ float  Sbuf[16][kNPad];    // 38 KB
  __shared__ __bf16 Pbuf[16][kNPad];    // 19 KB
  __shared__ float  redm[16][8];
  __shared__ float  reds[16][8];

  const int it = blockIdx.x;            // 16-row tile of queries (0..36)
  const int bh = blockIdx.y;
  const int b = bh / kH, h = bh % kH;
  const int tid  = threadIdx.x;
  const int lane = tid & 31;
  const int wave = tid >> 5;            // 4 waves
  const int m = lane & 15, g = lane >> 4;

  // ---- Phase 1: S tile-row = q @ k^T, waves split j-tiles (pipelined)
  {
    const __bf16* qrow = q + (bh * kNPad + it * 16 + m) * kD;
    v16bf qa0 = load_a_frag(qrow + 8 * g);        // K = d in [0,32)
    v16bf qa1 = load_a_frag(qrow + 32 + 8 * g);   // K = d in [32,64)
    const __bf16* kbase = kk + (bh * kNPad + m) * kD + 8 * g;

    int jt = wave;
    v16bf kb0 = load_a_frag(kbase + (jt * 16) * kD);
    v16bf kb1 = load_a_frag(kbase + (jt * 16) * kD + 32);
    while (jt < kJT) {
      const int jtn = (jt + 4 < kJT) ? jt + 4 : jt;
      v16bf nk0 = load_a_frag(kbase + (jtn * 16) * kD);
      v16bf nk1 = load_a_frag(kbase + (jtn * 16) * kD + 32);
      v8f s = {};
      s = wmma_bf16(qa0, kb0, s);
      s = wmma_bf16(qa1, kb1, s);
#pragma unroll
      for (int r = 0; r < 8; ++r) Sbuf[8 * g + r][jt * 16 + m] = s[r];
      kb0 = nk0; kb1 = nk1;
      jt += 4;
    }
  }
  __syncthreads();

  // ---- Phase 2: masked softmax in LDS (8 threads per row)
  {
    const int row = tid >> 3, sub = tid & 7;
    const int ig = it * 16 + row;                 // global query index
    const float sc = scale[h];
    float lm = -3.0e38f;
    for (int j = sub; j < kNPad; j += 8) {
      bool masked = (j >= kN) || (ig >= 1 && j >= 1 && j == ig);
      float v = masked ? -3.0e38f : Sbuf[row][j] * sc;
      Sbuf[row][j] = v;
      lm = fmaxf(lm, v);
    }
    redm[row][sub] = lm;
    __syncthreads();
    float mx = redm[row][0];
#pragma unroll
    for (int t = 1; t < 8; ++t) mx = fmaxf(mx, redm[row][t]);
    float ls = 0.f;
    for (int j = sub; j < kNPad; j += 8) {
      float e = __expf(Sbuf[row][j] - mx);
      Sbuf[row][j] = e;
      ls += e;
    }
    reds[row][sub] = ls;
    __syncthreads();
    float sum = 0.f;
#pragma unroll
    for (int t = 0; t < 8; ++t) sum += reds[row][t];
    float inv = 1.0f / sum;
    for (int j = sub; j < kNPad; j += 8)
      Pbuf[row][j] = f2bf(Sbuf[row][j] * inv);
  }
  __syncthreads();

  // ---- Phase 3: out = P @ V, each wave owns one 16-wide d-tile (pipelined)
  {
    const int dt = wave;                          // 0..3 (4*16 = 64 = D)
    const __bf16* vbase = vT + (bh * kD + dt * 16 + m) * kNPad + 8 * g;
    v8f acc = {};
    v16bf pa = load_a_frag(&Pbuf[m][8 * g]);
    v16bf vb = load_a_frag(vbase);
    for (int kt = 0; kt < kKTpv; ++kt) {
      const int ktn = (kt + 1 < kKTpv) ? kt + 1 : kt;
      v16bf npa = load_a_frag(&Pbuf[m][ktn * 32 + 8 * g]);
      v16bf nvb = load_a_frag(vbase + ktn * 32);
      acc = wmma_bf16(pa, vb, acc);
      pa = npa; vb = nvb;
    }
#pragma unroll
    for (int r = 0; r < 8; ++r) {
      int n = it * 16 + 8 * g + r;
      if (n < kN)
        ao[(b * kN + n) * kC + h * kD + dt * 16 + m] = f2bf(acc[r]);
    }
  }
}

// ---------------------------------------------------------------- stage 3: output projection
// Per wave: 32(M) x 64(N) tile, K = 768, pipelined like stage 1.

__global__ void __launch_bounds__(256)
proj_gemm_kernel(const __bf16* __restrict__ ab, const __bf16* __restrict__ wf,
                 const float* __restrict__ bias, float* __restrict__ out) {
  const int lane = threadIdx.x & 31;
  const int wave = threadIdx.x >> 5;
  constexpr int colTiles64 = kC / 64;   // 12
  constexpr int ntilesB = kC / 16;      // 48
  constexpr int KT = kC / 32;           // 24
  const int tile = blockIdx.x * 8 + wave;
  if (tile >= kRowTiles32 * colTiles64) return;
  const int rt = tile / colTiles64;
  const int ct = tile % colTiles64;
  const int m = lane & 15, g = lane >> 4;

  const __bf16* arow0 = ab + (rt * 32 + m) * kC + 8 * g;
  const __bf16* arow1 = ab + (rt * 32 + 16 + m) * kC + 8 * g;
  const __bf16* bbase = wf + ((ct * 4) << 9) + (lane << 4);

  v8f acc[2][4];
#pragma unroll
  for (int i = 0; i < 2; ++i)
#pragma unroll
    for (int j = 0; j < 4; ++j) acc[i][j] = v8f{};

  v16bf a0 = load_a_frag(arow0);
  v16bf a1 = load_a_frag(arow1);
  v16bf b0 = load_b_frag(bbase);
  v16bf b1 = load_b_frag(bbase + 512);
  v16bf b2 = load_b_frag(bbase + 1024);
  v16bf b3 = load_b_frag(bbase + 1536);

  for (int kt = 0; kt < KT; ++kt) {
    const int ktn = (kt + 1 < KT) ? kt + 1 : kt;
    v16bf na0 = load_a_frag(arow0 + ktn * 32);
    v16bf na1 = load_a_frag(arow1 + ktn * 32);
    const __bf16* nb = bbase + ((ktn * ntilesB) << 9);
    v16bf nb0 = load_b_frag(nb);
    v16bf nb1 = load_b_frag(nb + 512);
    v16bf nb2 = load_b_frag(nb + 1024);
    v16bf nb3 = load_b_frag(nb + 1536);

    acc[0][0] = wmma_bf16(a0, b0, acc[0][0]);
    acc[0][1] = wmma_bf16(a0, b1, acc[0][1]);
    acc[0][2] = wmma_bf16(a0, b2, acc[0][2]);
    acc[0][3] = wmma_bf16(a0, b3, acc[0][3]);
    acc[1][0] = wmma_bf16(a1, b0, acc[1][0]);
    acc[1][1] = wmma_bf16(a1, b1, acc[1][1]);
    acc[1][2] = wmma_bf16(a1, b2, acc[1][2]);
    acc[1][3] = wmma_bf16(a1, b3, acc[1][3]);

    a0 = na0; a1 = na1; b0 = nb0; b1 = nb1; b2 = nb2; b3 = nb3;
  }

#pragma unroll
  for (int j = 0; j < 4; ++j) {
    const int c = ct * 64 + j * 16 + m;
    const float bv = bias[c];
#pragma unroll
    for (int i = 0; i < 2; ++i) {
#pragma unroll
      for (int r = 0; r < 8; ++r) {
        int row = rt * 32 + i * 16 + 8 * g + r;
        out[row * kC + c] = acc[i][j][r] + bv;
      }
    }
  }
}

// ---------------------------------------------------------------- launch

extern "C" void kernel_launch(void* const* d_in, const int* in_sizes, int n_in,
                              void* d_out, int out_size, void* d_ws, size_t ws_size,
                              hipStream_t stream) {
  (void)in_sizes; (void)n_in; (void)out_size; (void)ws_size;
  const float* x      = (const float*)d_in[0];
  const float* w_qkv  = (const float*)d_in[1];
  const float* b_qkv  = (const float*)d_in[2];
  const float* scale  = (const float*)d_in[3];
  const float* w_proj = (const float*)d_in[4];
  const float* b_proj = (const float*)d_in[5];
  float* out = (float*)d_out;

  char* ws = (char*)d_ws;
  size_t off = 0;
  auto alloc = [&](size_t bytes) {
    char* p = ws + off;
    off = (off + bytes + 255) & ~size_t(255);
    return p;
  };
  __bf16* xb     = (__bf16*)alloc((size_t)kRows * kC * 2);             // 28.3 MB
  __bf16* wqkvf  = (__bf16*)alloc((size_t)kC * kC3 * 2);               //  3.5 MB
  __bf16* wprojf = (__bf16*)alloc((size_t)kC * kC * 2);                //  1.2 MB
  __bf16* qb     = (__bf16*)alloc((size_t)kB * kH * kNPad * kD * 2);   // 29.9 MB
  __bf16* kb     = (__bf16*)alloc((size_t)kB * kH * kNPad * kD * 2);   // 29.9 MB
  __bf16* vTb    = (__bf16*)alloc((size_t)kB * kH * kD * kNPad * 2);   // 29.9 MB
  __bf16* ao     = (__bf16*)alloc((size_t)kRows * kC * 2);             // 28.3 MB

  cvt_f32_bf16_kernel<<<2048, 256, 0, stream>>>(x, xb, kRows * kC);
  swizzle_w_kernel<<<1024, 256, 0, stream>>>(w_qkv, wqkvf, kC3, kC * kC3);
  swizzle_w_kernel<<<512,  256, 0, stream>>>(w_proj, wprojf, kC, kC * kC);

  qkv_gemm_kernel<<<(kRowTiles32 * (kC3 / 64) + 7) / 8, 256, 0, stream>>>(
      xb, wqkvf, b_qkv, qb, kb, vTb);

  attn_kernel<<<dim3(37, kB * kH), 128, 0, stream>>>(qb, kb, vTb, scale, ao);

  proj_gemm_kernel<<<(kRowTiles32 * (kC / 64) + 7) / 8, 256, 0, stream>>>(
      ao, wprojf, b_proj, out);
}
